// DGCNN_80650895884582
// MI455X (gfx1250) — matrix operand order repaired and checked by reading, not compiled, not run on hardware
//
#include <hip/hip_runtime.h>
#include <hip/hip_bf16.h>

typedef __attribute__((ext_vector_type(16))) _Float16 v16h;
typedef __attribute__((ext_vector_type(8)))  _Float16 v8h;
typedef __attribute__((ext_vector_type(8)))  float    v8f;
typedef __attribute__((ext_vector_type(2)))  float    v2f;

#define KNBR 20
#define BATCH 8
#define NPTS 2048

// ---------------------------------------------------------------------------
// Convert f32 weights (O x Kin, row major) to f16 with K padded to KP (zeros).
// ---------------------------------------------------------------------------
__global__ void convert_w_kernel(const float* __restrict__ src,
                                 _Float16* __restrict__ dst,
                                 int O, int Kin, int KP) {
    int i = blockIdx.x * blockDim.x + threadIdx.x;
    int tot = O * KP;
    if (i < tot) {
        int o = i / KP, c = i % KP;
        dst[i] = (c < Kin) ? (_Float16)src[o * Kin + c] : (_Float16)0.0f;
    }
}

// ---------------------------------------------------------------------------
// Squared norms per point: xx[b*N+n] = sum_c x[b,n,c]^2
// ---------------------------------------------------------------------------
__global__ void norms_kernel(const float* __restrict__ x, float* __restrict__ xx, int C) {
    int i = blockIdx.x * blockDim.x + threadIdx.x;
    if (i < BATCH * NPTS) {
        float s = 0.f;
        const float* p = x + (size_t)i * C;
        for (int c = 0; c < C; ++c) s += p[c] * p[c];
        xx[i] = s;
    }
}

// ---------------------------------------------------------------------------
// kNN via f32 WMMA Gram tiles (full f32 precision so the discrete top-k
// selection matches the f32 reference).
//   Block = 16 queries, 4 waves. Per stripe each wave computes a
//   16(cand) x 16(query) inner-product tile with V_WMMA_F32_16X16X4_F32
//   (A = candidate features, B = query features, both staged in LDS).
//   dist = 2*dot - qq - cc feeds per-lane register top-20 insertion nets;
//   8 partial lists per query merge through LDS at the end.
// ---------------------------------------------------------------------------
template <int C>
__global__ void knn_kernel(const float* __restrict__ x, const float* __restrict__ xx,
                           int* __restrict__ idx) {
    constexpr int CP  = (C < 4) ? 4 : C;   // pad K to multiple of 4
    constexpr int NCH = CP / 4;            // WMMA K chunks
    const int b    = blockIdx.y;
    const int q0   = blockIdx.x * 16;
    const int tid  = threadIdx.x;
    const int lane = tid & 31, wave = tid >> 5;

    __shared__ alignas(16) float qf[16][CP];
    __shared__ float qq[16];
    __shared__ alignas(16) float cand[4][16][CP];
    __shared__ float cc[4][16];
    __shared__ float md[16][8 * KNBR];
    __shared__ int   mi[16][8 * KNBR];

    const float* xb  = x  + (size_t)b * NPTS * C;
    const float* xxb = xx + (size_t)b * NPTS;

    for (int i = tid; i < 16 * CP; i += 128) {
        int m = i / CP, c = i % CP;
        qf[m][c] = (c < C) ? xb[(size_t)(q0 + m) * C + c] : 0.f;
    }
    if (tid < 16) qq[tid] = xxb[q0 + tid];

    const int q    = lane & 15;            // query column of this lane
    const int half = lane >> 4;
    const int kb2  = half * 2;

    float bd[KNBR]; int bi[KNBR];
#pragma unroll
    for (int j = 0; j < KNBR; ++j) { bd[j] = -__builtin_inff(); bi[j] = 0x7fffffff; }

    for (int it = 0; it < NPTS / 64; ++it) {
        const int cbase = it * 64 + wave * 16;
        __syncthreads();                                   // protect cand/cc
        for (int i = lane; i < 16 * CP; i += 32) {
            int m = i / CP, c = i % CP;
            cand[wave][m][c] = (c < C) ? xb[(size_t)(cbase + m) * C + c] : 0.f;
        }
        if (lane < 16) cc[wave][lane] = xxb[cbase + lane];
        __syncthreads();

        v8f acc = {};
#pragma unroll
        for (int t = 0; t < NCH; ++t) {
            v2f a  = *(const v2f*)(&cand[wave][q][4 * t + kb2]);  // A: cand row
            v2f bq = *(const v2f*)(&qf[q][4 * t + kb2]);          // B: query col
            acc = __builtin_amdgcn_wmma_f32_16x16x4_f32(
                false, a, false, bq, (short)0, acc, false, false);
        }

        const float qqv = qq[q];
#pragma unroll
        for (int r = 0; r < 8; ++r) {
            int cl = r + 8 * half;                         // candidate row in tile
            float d = 2.f * acc[r] - qqv - cc[wave][cl];
            int   ci = cbase + cl;
#pragma unroll
            for (int j = 0; j < KNBR; ++j) {
                bool better = (d > bd[j]) || (d == bd[j] && ci < bi[j]);
                float td = bd[j]; int ti = bi[j];
                if (better) { bd[j] = d; bi[j] = ci; d = td; ci = ti; }
            }
        }
    }

    // Dump the 8 partial lists per query, then merge.
    __syncthreads();
    const int slot = wave * 2 + half;
#pragma unroll
    for (int j = 0; j < KNBR; ++j) {
        md[q][slot * KNBR + j] = bd[j];
        mi[q][slot * KNBR + j] = bi[j];
    }
    __syncthreads();

    if (tid < 16) {
        int heads[8];
        for (int l = 0; l < 8; ++l) heads[l] = 0;
        int* op = idx + ((size_t)b * NPTS + q0 + tid) * KNBR;
        for (int j = 0; j < KNBR; ++j) {
            float best = -__builtin_inff(); int bidx = 0x7fffffff; int bl = 0;
            for (int l = 0; l < 8; ++l) {
                int h = heads[l];
                if (h >= KNBR) continue;
                float dd = md[tid][l * KNBR + h];
                int   ii = mi[tid][l * KNBR + h];
                if (dd > best || (dd == best && ii < bidx)) { best = dd; bidx = ii; bl = l; }
            }
            op[j] = bidx;
            heads[bl]++;
        }
    }
}

// ---------------------------------------------------------------------------
// Fused EdgeConv: out[b,n,o] = max_k leaky( [nbr-x | x] @ W[o,:] )
// Block = 16 nodes, COUT/16 waves (each wave owns a 16-column slice).
// ---------------------------------------------------------------------------
template <int CIN, int COUT>
__global__ void edge_conv_kernel(const float* __restrict__ x,
                                 const _Float16* __restrict__ wt,
                                 const int* __restrict__ idx,
                                 float* __restrict__ out) {
    constexpr int KP = ((2 * CIN) + 31) / 32 * 32;
    constexpr int NC = KP / 32;
    constexpr int NW = COUT / 16;
    const int b   = blockIdx.y;
    const int n0  = blockIdx.x * 16;
    const int tid = threadIdx.x;
    const int lane = tid & 31, wave = tid >> 5;
    const int nth = NW * 32;

    __shared__ float xc[16][CIN];
    __shared__ float nb[16][CIN];
    __shared__ alignas(64) _Float16 ah[16][KP];
    __shared__ int nid[16];

    const float* xb = x + (size_t)b * NPTS * CIN;

    for (int i = tid; i < 16 * CIN; i += nth) {
        int m = i / CIN, c = i % CIN;
        xc[m][c] = xb[(size_t)(n0 + m) * CIN + c];
    }

    const int col  = 16 * wave + (lane & 15);
    const int kb16 = (lane >> 4) * 16;
    v16h bfr[NC];
#pragma unroll
    for (int kc = 0; kc < NC; ++kc)
        bfr[kc] = *(const v16h*)(wt + (size_t)col * KP + kc * 32 + kb16);

    float outv[8];
#pragma unroll
    for (int r = 0; r < 8; ++r) outv[r] = -__builtin_inff();

    const int row = lane & 15;
    const int kb8 = (lane >> 4) * 16;

    for (int k = 0; k < KNBR; ++k) {
        __syncthreads();
        if (tid < 16)
            nid[tid] = idx[((size_t)b * NPTS + n0 + tid) * KNBR + k];
        __syncthreads();
        for (int i = tid; i < 16 * CIN; i += nth) {
            int m = i / CIN, c = i % CIN;
            nb[m][c] = xb[(size_t)nid[m] * CIN + c];
        }
        __syncthreads();
        for (int i = tid; i < 16 * KP; i += nth) {
            int m = i / KP, c = i % KP;
            float v;
            if      (c < CIN)     v = nb[m][c] - xc[m][c];
            else if (c < 2 * CIN) v = xc[m][c - CIN];
            else                  v = 0.0f;
            ah[m][c] = (_Float16)v;
        }
        __syncthreads();

        v8f acc = {};
#pragma unroll
        for (int kc = 0; kc < NC; ++kc) {
            v8h lo = *(const v8h*)(&ah[row][kc * 32 + kb8]);
            v8h hi = *(const v8h*)(&ah[row][kc * 32 + kb8 + 8]);
            v16h a;
#pragma unroll
            for (int e = 0; e < 8; ++e) { a[e] = lo[e]; a[8 + e] = hi[e]; }
            acc = __builtin_amdgcn_wmma_f32_16x16x32_f16(
                false, a, false, bfr[kc], (short)0, acc, false, false);
        }
#pragma unroll
        for (int r = 0; r < 8; ++r) {
            float h = acc[r];
            h = h > 0.f ? h : 0.2f * h;
            outv[r] = fmaxf(outv[r], h);
        }
    }

    float* ob = out + (size_t)b * NPTS * COUT;
#pragma unroll
    for (int r = 0; r < 8; ++r) {
        int m = r + 8 * (lane >> 4);
        ob[(size_t)(n0 + m) * COUT + col] = outv[r];
    }
}

// ---------------------------------------------------------------------------
// Final: out = leaky( [x1|x2|x3|x4] @ W5^T ), 16 nodes/block, 8 waves, K=512.
// ---------------------------------------------------------------------------
__global__ void final_kernel(const float* __restrict__ x1, const float* __restrict__ x2,
                             const float* __restrict__ x3, const float* __restrict__ x4,
                             const _Float16* __restrict__ wt, float* __restrict__ out) {
    constexpr int KP = 512, NC = 16, COUT = 128, NW = 8;
    const int b   = blockIdx.y;
    const int n0  = blockIdx.x * 16;
    const int tid = threadIdx.x;
    const int lane = tid & 31, wave = tid >> 5;
    const int nth = NW * 32;

    __shared__ alignas(64) _Float16 ah[16][KP];

    for (int i = tid; i < 16 * KP; i += nth) {
        int m = i / KP, c = i % KP;
        size_t n = (size_t)b * NPTS + n0 + m;
        float v;
        if      (c < 64)  v = x1[n * 64  + c];
        else if (c < 128) v = x2[n * 64  + (c - 64)];
        else if (c < 256) v = x3[n * 128 + (c - 128)];
        else              v = x4[n * 256 + (c - 256)];
        ah[m][c] = (_Float16)v;
    }
    __syncthreads();

    const int col  = 16 * wave + (lane & 15);
    const int kb16 = (lane >> 4) * 16;
    const int row  = lane & 15;
    const int kb8  = (lane >> 4) * 16;

    v8f acc = {};
#pragma unroll
    for (int kc = 0; kc < NC; ++kc) {
        v8h lo = *(const v8h*)(&ah[row][kc * 32 + kb8]);
        v8h hi = *(const v8h*)(&ah[row][kc * 32 + kb8 + 8]);
        v16h a;
#pragma unroll
        for (int e = 0; e < 8; ++e) { a[e] = lo[e]; a[8 + e] = hi[e]; }
        v16h bv = *(const v16h*)(wt + (size_t)col * KP + kc * 32 + kb16);
        acc = __builtin_amdgcn_wmma_f32_16x16x32_f16(
            false, a, false, bv, (short)0, acc, false, false);
    }

    float* ob = out + (size_t)b * NPTS * COUT;
#pragma unroll
    for (int r = 0; r < 8; ++r) {
        float h = acc[r];
        h = h > 0.f ? h : 0.2f * h;
        int m = r + 8 * (lane >> 4);
        ob[(size_t)(n0 + m) * COUT + col] = h;
    }
}

// ---------------------------------------------------------------------------
extern "C" void kernel_launch(void* const* d_in, const int* in_sizes, int n_in,
                              void* d_out, int out_size, void* d_ws, size_t ws_size,
                              hipStream_t stream) {
    (void)in_sizes; (void)n_in; (void)out_size; (void)ws_size;
    const float* x  = (const float*)d_in[0];   // (8,2048,3)
    const float* W1 = (const float*)d_in[1];   // (64,6)
    const float* W2 = (const float*)d_in[2];   // (64,128)
    const float* W3 = (const float*)d_in[3];   // (128,128)
    const float* W4 = (const float*)d_in[4];   // (256,256)
    const float* W5 = (const float*)d_in[5];   // (128,512)
    float* out = (float*)d_out;                // (8,2048,128)

    char*  base = (char*)d_ws;
    size_t off  = 0;
    auto carve = [&](size_t bytes) -> char* {
        char* p = base + off;
        off = (off + bytes + 255) & ~(size_t)255;
        return p;
    };
    _Float16* w1h = (_Float16*)carve((size_t)64  * 32  * 2);
    _Float16* w2h = (_Float16*)carve((size_t)64  * 128 * 2);
    _Float16* w3h = (_Float16*)carve((size_t)128 * 128 * 2);
    _Float16* w4h = (_Float16*)carve((size_t)256 * 256 * 2);
    _Float16* w5h = (_Float16*)carve((size_t)128 * 512 * 2);
    int*   idx = (int*)  carve((size_t)BATCH * NPTS * KNBR * 4);
    float* xx  = (float*)carve((size_t)BATCH * NPTS * 4);
    float* x1  = (float*)carve((size_t)BATCH * NPTS * 64  * 4);
    float* x2  = (float*)carve((size_t)BATCH * NPTS * 64  * 4);
    float* x3  = (float*)carve((size_t)BATCH * NPTS * 128 * 4);
    float* x4  = (float*)carve((size_t)BATCH * NPTS * 256 * 4);

    auto cvt = [&](const float* s, _Float16* d, int O, int Kin, int KP) {
        int tot = O * KP;
        convert_w_kernel<<<(tot + 255) / 256, 256, 0, stream>>>(s, d, O, Kin, KP);
    };
    cvt(W1, w1h, 64, 6, 32);
    cvt(W2, w2h, 64, 128, 128);
    cvt(W3, w3h, 128, 128, 128);
    cvt(W4, w4h, 256, 256, 256);
    cvt(W5, w5h, 128, 512, 512);

    dim3 kgrid(NPTS / 16, BATCH);
    dim3 egrid(NPTS / 16, BATCH);
    int  ngrid = (BATCH * NPTS + 255) / 256;

    // Layer 1: kNN on xyz, EdgeConv 3->64 (K padded 6->32).
    norms_kernel<<<ngrid, 256, 0, stream>>>(x, xx, 3);
    knn_kernel<3><<<kgrid, 128, 0, stream>>>(x, xx, idx);
    edge_conv_kernel<3, 64><<<egrid, 4 * 32, 0, stream>>>(x, w1h, idx, x1);

    // Layer 2: kNN on x1 (64), EdgeConv 64->64.
    norms_kernel<<<ngrid, 256, 0, stream>>>(x1, xx, 64);
    knn_kernel<64><<<kgrid, 128, 0, stream>>>(x1, xx, idx);
    edge_conv_kernel<64, 64><<<egrid, 4 * 32, 0, stream>>>(x1, w2h, idx, x2);

    // Layer 3: kNN on x2 (64), EdgeConv 64->128.
    norms_kernel<<<ngrid, 256, 0, stream>>>(x2, xx, 64);
    knn_kernel<64><<<kgrid, 128, 0, stream>>>(x2, xx, idx);
    edge_conv_kernel<64, 128><<<egrid, 8 * 32, 0, stream>>>(x2, w3h, idx, x3);

    // Layer 4: kNN on x3 (128), EdgeConv 128->256.
    norms_kernel<<<ngrid, 256, 0, stream>>>(x3, xx, 128);
    knn_kernel<128><<<kgrid, 128, 0, stream>>>(x3, xx, idx);
    edge_conv_kernel<128, 256><<<egrid, 16 * 32, 0, stream>>>(x3, w4h, idx, x4);

    // Final pointwise linear over the 512-wide concat.
    final_kernel<<<egrid, 8 * 32, 0, stream>>>(x1, x2, x3, x4, w5h, out);
}
